// CausalAttentation_63367947485916
// MI455X (gfx1250) — matrix operand order repaired and checked by reading, not compiled, not run on hardware
//
#include <hip/hip_runtime.h>

// ---------------------------------------------------------------------------
// Causal single-head attention, MI455X (gfx1250, wave32, WMMA bf16 path).
//   q,k,v = x @ W^T ; S = qk^T/sqrt(d) causal ; out = softmax(S) @ v
// Shapes: x[4,4096,1024], W[1024,1024], out[4,4096,1024] (f32 in/out).
// All matmuls via v_wmma_f32_16x16x32_bf16 (f32 accumulate).
// ---------------------------------------------------------------------------

#define BATCH 4
#define SEQ   4096
#define DM    1024   // d_in == d_out

typedef __bf16 bf16_t;
typedef __attribute__((ext_vector_type(16))) __bf16 v16bf;
typedef __attribute__((ext_vector_type(8)))  __bf16 v8bf;
typedef __attribute__((ext_vector_type(8)))  float  v8f;

static __device__ __forceinline__ v8f wmma_bf16(v16bf a, v16bf b, v8f c) {
  // 8 args: (neg_a, A, neg_b, B, c_mod, C, reuse_a, reuse_b)
  return __builtin_amdgcn_wmma_f32_16x16x32_bf16(false, a, false, b, (short)0, c,
                                                 false, false);
}

// Assemble a 16-bit A operand (16x32): lane holds row M=lane%16;
// vector elems 0..15 -> K = g*16+e  (g = lane/16).
// 'p' must point at row start + kBase + g*16 (two contiguous 8xbf16 chunks, +8 apart).
static __device__ __forceinline__ v16bf load_a16x32(const bf16_t* p) {
  v8bf lo = *(const v8bf*)p;
  v8bf hi = *(const v8bf*)(p + 8);
  v16bf a;
#pragma unroll
  for (int i = 0; i < 8; ++i) { a[i] = lo[i]; a[i + 8] = hi[i]; }
  return a;
}

// ---------------------------------------------------------------------------
// Stage 1: f32 -> bf16 convert, 8 elems/thread (n must be a multiple of 8).
// ---------------------------------------------------------------------------
__global__ void cvt_f32_bf16_x8(const float* __restrict__ s,
                                bf16_t* __restrict__ d, int n) {
  int i = (blockIdx.x * blockDim.x + threadIdx.x) * 8;
  if (i + 8 > n) return;
  float4 a = *(const float4*)(s + i);
  float4 b = *(const float4*)(s + i + 4);
  v8bf o;
  o[0] = (bf16_t)a.x; o[1] = (bf16_t)a.y; o[2] = (bf16_t)a.z; o[3] = (bf16_t)a.w;
  o[4] = (bf16_t)b.x; o[5] = (bf16_t)b.y; o[6] = (bf16_t)b.z; o[7] = (bf16_t)b.w;
  *(v8bf*)(d + i) = o;
}

// ---------------------------------------------------------------------------
// Stage 2: QKV projection GEMM.  out[r,c] = sum_k X[r,k] * W[c,k]
// grid = (DM/64, (B*SEQ)/(4*32), 3), block = 128 (4 waves).
// Wave tile: M=32, N=64 -> 8 f32 accumulators, 8 WMMAs per K-step of 32.
// z==2 (V) is stored transposed: Vt[b][d][t] so PV B-loads are contiguous.
// The z branch is hoisted OUTSIDE the store loops (z is block-uniform).
// ---------------------------------------------------------------------------
__global__ __launch_bounds__(128)
void qkv_gemm(const bf16_t* __restrict__ X,
              const bf16_t* __restrict__ Wq, const bf16_t* __restrict__ Wk,
              const bf16_t* __restrict__ Wv,
              bf16_t* __restrict__ Q, bf16_t* __restrict__ K,
              bf16_t* __restrict__ Vt) {
  const int lane = threadIdx.x & 31;
  const int wave = threadIdx.x >> 5;
  const int g    = lane >> 4;        // lane group (0/1)
  const int ln   = lane & 15;
  const int rBase = (blockIdx.y * 4 + wave) * 32;
  const int cBase = blockIdx.x * 64;
  const int z     = blockIdx.z;
  const bf16_t* W = (z == 0) ? Wq : ((z == 1) ? Wk : Wv);

  const v8f vzero = {};
  v8f acc[2][4];
#pragma unroll
  for (int mt = 0; mt < 2; ++mt)
#pragma unroll
    for (int nt = 0; nt < 4; ++nt) acc[mt][nt] = vzero;

  for (int k0 = 0; k0 < DM; k0 += 32) {
    v16bf A[2];
#pragma unroll
    for (int mt = 0; mt < 2; ++mt)
      A[mt] = load_a16x32(X + (size_t)(rBase + mt * 16 + ln) * DM + k0 + g * 16);
    v16bf Bm[4];
#pragma unroll
    for (int nt = 0; nt < 4; ++nt)
      // B operand (K=32 x N=16): lane = col N, 16 contiguous K per lane group.
      Bm[nt] = *(const v16bf*)(W + (size_t)(cBase + nt * 16 + ln) * DM + k0 + g * 16);
#pragma unroll
    for (int mt = 0; mt < 2; ++mt)
#pragma unroll
      for (int nt = 0; nt < 4; ++nt)
        acc[mt][nt] = wmma_bf16(A[mt], Bm[nt], acc[mt][nt]);
  }

  // ---- epilogue: single uniform branch, then straight-line store loops ----
  if (z == 2) {
#pragma unroll
    for (int mt = 0; mt < 2; ++mt)
#pragma unroll
      for (int nt = 0; nt < 4; ++nt)
#pragma unroll
        for (int r = 0; r < 8; ++r) {
          const int row = rBase + mt * 16 + g * 8 + r;   // C-tile: M = r + 8*g
          const int col = cBase + nt * 16 + ln;          //         N = lane%16
          const int b = row >> 12, t = row & (SEQ - 1);
          Vt[((size_t)b * DM + col) * SEQ + t] = (bf16_t)acc[mt][nt][r];
        }
  } else {
    bf16_t* __restrict__ O = (z == 0) ? Q : K;
#pragma unroll
    for (int mt = 0; mt < 2; ++mt)
#pragma unroll
      for (int nt = 0; nt < 4; ++nt)
#pragma unroll
        for (int r = 0; r < 8; ++r) {
          const int row = rBase + mt * 16 + g * 8 + r;
          const int col = cBase + nt * 16 + ln;
          O[(size_t)row * DM + col] = (bf16_t)acc[mt][nt][r];
        }
  }
}

// ---------------------------------------------------------------------------
// Stage 3: fused causal flash attention.
// grid = (SEQ/16, BATCH), block = 256 (8 waves).
// Block owns 16 query rows; wave w owns output d-chunk [128w, 128w+128).
// Per 32-key step: partial 16x32 S per wave (8 WMMAs over its d-chunk),
// cross-wave LDS reduction, online softmax (shfl_xor row reduce),
// P -> LDS (C-layout -> A-layout), O += P @ V (8 WMMAs).
// ---------------------------------------------------------------------------
__global__ __launch_bounds__(256)
void flash_attn(const bf16_t* __restrict__ Q, const bf16_t* __restrict__ Kc,
                const bf16_t* __restrict__ Vt, float* __restrict__ out) {
  __shared__ float  red[8 * 512];   // per-wave partial S (2 tiles x 256)
  __shared__ float  sred[512];      // reduced S
  __shared__ bf16_t pl[16 * 32];    // P in row-major (16 rows x 32 keys)

  const int lane  = threadIdx.x & 31;
  const int wave  = threadIdx.x >> 5;
  const int tid   = threadIdx.x;
  const int g     = lane >> 4;
  const int ln    = lane & 15;
  const int qBase = blockIdx.x * 16;
  const int b     = blockIdx.y;
  const int dBase = wave * 128;

  // Q A-tiles for this wave's d-chunk (4 tiles of 16x32), kept in VGPRs.
  v16bf qa[4];
#pragma unroll
  for (int c = 0; c < 4; ++c)
    qa[c] = load_a16x32(Q + (size_t)(b * SEQ + qBase + ln) * DM + dBase + c * 32 + g * 16);

  const v8f vzero = {};
  v8f o[8];
#pragma unroll
  for (int t = 0; t < 8; ++t) o[t] = vzero;
  float m[8], l[8];
#pragma unroll
  for (int r = 0; r < 8; ++r) { m[r] = -1e30f; l[r] = 0.f; }

  const int nkb = (qBase >> 5) + 1;            // 32-key blocks up to diagonal
  for (int kb = 0; kb < nkb; ++kb) {
    const int jBase = kb * 32;

    // --- partial S = Q K^T over this wave's d-chunk (two 16x16 col tiles) ---
    v8f s0 = vzero, s1 = vzero;
    const bf16_t* kp = Kc + (size_t)(b * SEQ + jBase + ln) * DM + dBase + g * 16;
#pragma unroll
    for (int c = 0; c < 4; ++c) {
      v16bf b0 = *(const v16bf*)(kp + c * 32);                   // keys jBase+ln
      v16bf b1 = *(const v16bf*)(kp + (size_t)16 * DM + c * 32); // keys +16
      s0 = wmma_bf16(qa[c], b0, s0);
      s1 = wmma_bf16(qa[c], b1, s1);
    }

    // --- cross-wave reduction of partial S through LDS ---
    __syncthreads();                               // protect red[] / pl[] reuse
#pragma unroll
    for (int r = 0; r < 8; ++r) {
      red[wave * 512 +       r * 32 + lane] = s0[r];
      red[wave * 512 + 256 + r * 32 + lane] = s1[r];
    }
    __syncthreads();
    {
      float a0 = 0.f, a1 = 0.f;
#pragma unroll
      for (int w = 0; w < 8; ++w) {
        a0 += red[w * 512 + tid];
        a1 += red[w * 512 + 256 + tid];
      }
      sred[tid] = a0; sred[256 + tid] = a1;
    }
    __syncthreads();

    // --- online softmax (scale 1/sqrt(1024)=1/32, causal mask) ---
    float p0[8], p1[8];
#pragma unroll
    for (int r = 0; r < 8; ++r) {
      float v0 = sred[r * 32 + lane] * 0.03125f;
      float v1 = sred[256 + r * 32 + lane] * 0.03125f;
      const int row = qBase + g * 8 + r;
      if (jBase + ln      > row) v0 = -1e30f;      // strict upper tri -> -inf
      if (jBase + 16 + ln > row) v1 = -1e30f;
      float rm = fmaxf(v0, v1);                    // row-max over 16 lanes
      rm = fmaxf(rm, __shfl_xor(rm, 1, 32));
      rm = fmaxf(rm, __shfl_xor(rm, 2, 32));
      rm = fmaxf(rm, __shfl_xor(rm, 4, 32));
      rm = fmaxf(rm, __shfl_xor(rm, 8, 32));
      const float mn = fmaxf(m[r], rm);
      p0[r] = __expf(v0 - mn);
      p1[r] = __expf(v1 - mn);
      float rs = p0[r] + p1[r];
      rs += __shfl_xor(rs, 1, 32);
      rs += __shfl_xor(rs, 2, 32);
      rs += __shfl_xor(rs, 4, 32);
      rs += __shfl_xor(rs, 8, 32);
      const float alpha = __expf(m[r] - mn);
      l[r] = l[r] * alpha + rs;
      m[r] = mn;
#pragma unroll
      for (int t = 0; t < 8; ++t) o[t][r] *= alpha;   // rescale O rows
    }

    // --- P: C-layout regs -> LDS row-major (identical in every wave) ---
    if (wave == 0) {
#pragma unroll
      for (int r = 0; r < 8; ++r) {
        const int row = g * 8 + r;
        pl[row * 32 + ln]      = (bf16_t)p0[r];
        pl[row * 32 + 16 + ln] = (bf16_t)p1[r];
      }
    }
    __syncthreads();

    // --- read P as A operand (16x32), then O += P @ V over d-chunk ---
    v16bf pa;
#pragma unroll
    for (int e = 0; e < 8; ++e) {
      pa[e]     = pl[ln * 32 + g * 16 + e];
      pa[e + 8] = pl[ln * 32 + g * 16 + 8 + e];
    }
#pragma unroll
    for (int t = 0; t < 8; ++t) {
      // B operand: B[k][n] = V[jBase+k][d] = Vt[b][d][jBase+k] (contiguous)
      const bf16_t* vp =
          Vt + ((size_t)b * DM + dBase + t * 16 + ln) * SEQ + jBase + g * 16;
      v16bf vb = *(const v16bf*)vp;
      o[t] = wmma_bf16(pa, vb, o[t]);
    }
  }

  // --- epilogue: normalize by softmax denominator (1 rcp per row), write f32 ---
  float inv[8];
#pragma unroll
  for (int r = 0; r < 8; ++r) inv[r] = 1.0f / l[r];
#pragma unroll
  for (int t = 0; t < 8; ++t)
#pragma unroll
    for (int r = 0; r < 8; ++r) {
      const int row = qBase + g * 8 + r;
      const int col = dBase + t * 16 + ln;
      out[((size_t)b * SEQ + row) * DM + col] = o[t][r] * inv[r];
    }
}

// ---------------------------------------------------------------------------
// Host launcher. Workspace layout (bytes):
//   Xb 32MB | Wqb 2MB | Wkb 2MB | Wvb 2MB | Qb 32MB | Kb 32MB | Vt 32MB = 134MB
// ---------------------------------------------------------------------------
extern "C" void kernel_launch(void* const* d_in, const int* in_sizes, int n_in,
                              void* d_out, int out_size, void* d_ws, size_t ws_size,
                              hipStream_t stream) {
  const float* x  = (const float*)d_in[0];
  const float* Wq = (const float*)d_in[1];
  const float* Wk = (const float*)d_in[2];
  const float* Wv = (const float*)d_in[3];
  float* out = (float*)d_out;

  char* ws = (char*)d_ws;
  const size_t NX = (size_t)BATCH * SEQ * DM;   // 16,777,216
  const size_t NW = (size_t)DM * DM;            // 1,048,576
  bf16_t* Xb  = (bf16_t*)ws;
  bf16_t* Wqb = (bf16_t*)(ws + NX * 2);
  bf16_t* Wkb = Wqb + NW;
  bf16_t* Wvb = Wkb + NW;
  bf16_t* Qb  = Wvb + NW;
  bf16_t* Kb  = Qb + NX;
  bf16_t* Vt  = Kb + NX;

  // Stage 1: f32 -> bf16
  cvt_f32_bf16_x8<<<(int)(NX / 8 / 256), 256, 0, stream>>>(x,  Xb,  (int)NX);
  cvt_f32_bf16_x8<<<(int)(NW / 8 / 256), 256, 0, stream>>>(Wq, Wqb, (int)NW);
  cvt_f32_bf16_x8<<<(int)(NW / 8 / 256), 256, 0, stream>>>(Wk, Wkb, (int)NW);
  cvt_f32_bf16_x8<<<(int)(NW / 8 / 256), 256, 0, stream>>>(Wv, Wvb, (int)NW);

  // Stage 2: Q/K/V projections (V stored transposed)
  dim3 gg(DM / 64, (BATCH * SEQ) / (4 * 32), 3);
  qkv_gemm<<<gg, 128, 0, stream>>>(Xb, Wqb, Wkb, Wvb, Qb, Kb, Vt);

  // Stage 3: fused causal flash attention
  dim3 ga(SEQ / 16, BATCH);
  flash_attn<<<ga, 256, 0, stream>>>(Qb, Kb, Vt, out);
}
